// Low_Decoder_111669150198
// MI455X (gfx1250) — matrix operand order repaired and checked by reading, not compiled, not run on hardware
//
#include <hip/hip_runtime.h>
#include <hip/hip_bf16.h>
#include <stdint.h>

typedef __attribute__((ext_vector_type(16))) _Float16 v16h;
typedef __attribute__((ext_vector_type(8)))  float    v8f;
typedef int v4i_vs __attribute__((vector_size(16)));   // matches builtin param type

#define BB 256   // batch
#define NN 128   // nodes / steps
#define EE 256   // embedding
#define HH 256   // hidden

// GEMM tiling: 4-wave workgroup, 64(M) x 64(N) block tile, K-slab 32
#define TM 64
#define TN 64
#define TK 32

// output layout (floats, concatenated in reference return order)
#define OUT_LOGP  0
#define OUT_IDX   (BB * NN)
#define OUT_INITN (2 * BB * NN)
#define OUT_LASTN (2 * BB * NN + 2 * BB)
#define OUT_RS    (2 * BB * NN + 4 * BB)

#if defined(__has_builtin)
#if __has_builtin(__builtin_amdgcn_global_load_async_to_lds_b128) && \
    __has_builtin(__builtin_amdgcn_s_wait_asynccnt)
#define USE_ASYNC_LDS 1
#endif
#endif

__device__ __forceinline__ uint64_t splitmix64(uint64_t x) {
  x += 0x9E3779B97F4A7C15ULL;
  x = (x ^ (x >> 30)) * 0xBF58476D1CE4E5B9ULL;
  x = (x ^ (x >> 27)) * 0x94D049BB133111EBULL;
  return x ^ (x >> 31);
}

// Generic GEMM: out[m,n] = sum_k A(m,k) * W[k,n] (+bias[n]) (+addMat[m,n])
// A split across two row-major buffers: k < K0 -> A0 (lda=K0), else A1 (lda=K1).
// 4 waves/block; each wave owns a 16-row strip and reuses its A fragment over
// four v_wmma_f32_16x16x32_f16 (16x64 of output). The W k-slab (32x64 f32) is
// staged in LDS once per block via async global->LDS copies.
// trans_seq > 0: store transposed per-sequence: out[(m/seq)*N + n][m%seq]
__global__ __launch_bounds__(128)
void wmma_gemm(const float* __restrict__ A0, int K0,
               const float* __restrict__ A1, int K1,
               const float* __restrict__ W,  int N,
               const float* __restrict__ bias,
               const float* __restrict__ addMat,
               float* __restrict__ out, int trans_seq)
{
  __shared__ float lds_w[TK * TN];           // 8 KB f32 W slab

  const int K    = K0 + K1;
  const int tid  = threadIdx.x;
  const int wave = tid >> 5;
  const int lane = tid & 31;
  const int half = lane >> 4;
  const int l    = lane & 15;
  const int m0   = blockIdx.x * TM + wave * 16;
  const int n0   = blockIdx.y * TN;
  const int m    = m0 + l;                   // A row owned by this lane

  v8f c0 = {}, c1 = {}, c2 = {}, c3 = {};

  for (int kb = 0; kb < K; kb += TK) {
    // ---- stage W[kb..kb+31, n0..n0+63] into LDS --------------------------
#pragma unroll
    for (int j = 0; j < 4; ++j) {
      const int f = (tid + 128 * j) * 4;     // flat float index, 16B per op
      const int r = f >> 6;                  // slab row (k offset)
      const int cc = f & 63;                 // slab col
      const float* gsrc = &W[(size_t)(kb + r) * N + (n0 + cc)];
#if defined(USE_ASYNC_LDS)
      __builtin_amdgcn_global_load_async_to_lds_b128(
          (__attribute__((address_space(1))) v4i_vs*)gsrc,
          (__attribute__((address_space(3))) v4i_vs*)&lds_w[f],
          0, 0);
#else
#pragma unroll
      for (int q = 0; q < 4; ++q) lds_w[f + q] = gsrc[q];
#endif
    }
#if defined(USE_ASYNC_LDS)
    __builtin_amdgcn_s_wait_asynccnt(0);
#endif
    __syncthreads();

    if (kb + TK < K)   // prefetch next W slab toward L2 while we compute
      __builtin_prefetch(&W[(size_t)(kb + TK) * N + n0 + (tid & 63)], 0, 1);

    // ---- A fragment (16-bit A 16x32 ISA layout) --------------------------
    v16h a;
#pragma unroll
    for (int i = 0; i < 16; ++i) {
      // lanes h=half: VGPR0-3 hold K = half*8 + 0..7, VGPR4-7 hold 16 + half*8 + 0..7
      const int k = kb + ((i < 8) ? (half * 8 + i) : (16 + half * 8 + (i - 8)));
      const float av = (k < K0) ? A0[(size_t)m * K0 + k]
                                : A1[(size_t)m * K1 + (k - K0)];
      a[i] = (_Float16)av;
    }

    // ---- 4 sub-tiles: B fragment from LDS, reuse A -----------------------
    // 16-bit B 32x16 layout (same k order as A): elems 0-7 K = half*8 + 0..7, elems 8-15 K = 16 + half*8 + 0..7
#pragma unroll
    for (int j = 0; j < 4; ++j) {
      v16h bf;
#pragma unroll
      for (int i = 0; i < 16; ++i)
        bf[i] = (_Float16)lds_w[((i < 8) ? (half * 8 + i) : (16 + half * 8 + (i - 8))) * TN + (j * 16 + l)];
      v8f acc = (j == 0) ? c0 : (j == 1) ? c1 : (j == 2) ? c2 : c3;
      acc = __builtin_amdgcn_wmma_f32_16x16x32_f16(false, a, false, bf,
                                                   (short)0, acc, false, false);
      if      (j == 0) c0 = acc;
      else if (j == 1) c1 = acc;
      else if (j == 2) c2 = acc;
      else             c3 = acc;
    }
    __syncthreads();   // protect slab before next-iteration overwrite
  }

  // ---- epilogue ----------------------------------------------------------
#pragma unroll
  for (int j = 0; j < 4; ++j) {
    const v8f acc = (j == 0) ? c0 : (j == 1) ? c1 : (j == 2) ? c2 : c3;
    const int n = n0 + j * 16 + l;           // C col owned by this lane
#pragma unroll
    for (int v = 0; v < 8; ++v) {
      const int mr = m0 + v + half * 8;      // C layout: VGPR v <-> M = v + 8*half
      float val = acc[v];
      if (bias)   val += bias[n];
      if (addMat) val += addMat[(size_t)mr * N + n];
      if (trans_seq > 0) {
        const int bb = mr / trans_seq;
        const int ns = mr % trans_seq;
        out[((size_t)bb * N + n) * trans_seq + ns] = val;
      } else {
        out[(size_t)mr * N + n] = val;
      }
    }
  }
}

// mean over the node axis: meanlcv[b,e] = (1/N) sum_n lcv[b,n,e]
__global__ __launch_bounds__(256)
void mean_lcv(const float* __restrict__ lcv, float* __restrict__ mlcv) {
  const int i = blockIdx.x * blockDim.x + threadIdx.x;   // 0..B*E-1
  const int b = i / EE, e = i % EE;
  float s = 0.f;
  for (int nn = 0; nn < NN; ++nn) s += lcv[((size_t)b * NN + nn) * EE + e];
  mlcv[i] = s * (1.0f / NN);
}

// rest0[e] = b_rest[e] + sum_k low_init_w[k] * W_rest[k,e]   (vector-matrix, tiny)
__global__ __launch_bounds__(EE)
void rest0_kernel(const float* __restrict__ liw, const float* __restrict__ W_rest,
                  const float* __restrict__ b_rest, float* __restrict__ rest0) {
  const int e = threadIdx.x;
  float s = b_rest[e];
  for (int k = 0; k < 2 * EE; ++k) s += liw[k] * W_rest[(size_t)k * EE + e];
  rest0[e] = s;
}

// query0 = h_bar + rest0; zero/copy state; emit init_node output
__global__ __launch_bounds__(256)
void init_state(const float* __restrict__ h_bar, const float* __restrict__ rest0,
                const int* __restrict__ mask_in, const float* __restrict__ onode,
                float* __restrict__ query, int* __restrict__ maskws,
                int* __restrict__ cur_idx, float* __restrict__ init_h,
                float* __restrict__ h_cur, float* __restrict__ out) {
  const int i = blockIdx.x * blockDim.x + threadIdx.x;   // 0..B*E-1
  query[i]  = h_bar[i] + rest0[i & (EE - 1)];
  init_h[i] = 0.f;
  h_cur[i]  = 0.f;
  if (i < BB * NN) maskws[i] = mask_in[i];
  if (i < BB) {
    cur_idx[i] = 0;
    out[OUT_INITN + i * 2 + 0] = onode[(size_t)i * NN * 2 + 0];
    out[OUT_INITN + i * 2 + 1] = onode[(size_t)i * NN * 2 + 1];
  }
}

// One step of the decode scan. One block per batch element, one thread per node.
__global__ __launch_bounds__(NN)
void attn_sample(const float* __restrict__ keysT, const float* __restrict__ qbuf,
                 const float* __restrict__ vvec,  const float* __restrict__ lcv,
                 const float* __restrict__ onode, const int* __restrict__ idptr,
                 int* __restrict__ maskws, int* __restrict__ cur_idx,
                 float* __restrict__ h_cur, float* __restrict__ init_h,
                 float* __restrict__ out, int step)
{
  __shared__ float sh_logit[NN];
  __shared__ float sh_red[NN];
  __shared__ int   sh_arg[NN];
  __shared__ int   s_sel;
  __shared__ float s_max;

  const int b = blockIdx.x;
  const int t = threadIdx.x;   // node index n

  const float* kb = keysT + (size_t)b * HH * NN;   // coalesced across threads
  const float* qb = qbuf  + (size_t)b * HH;        // uniform -> scalar loads
  float u = 0.f;
  for (int h = 0; h < HH; ++h)
    u = fmaf(tanhf(kb[h * NN + t] + qb[h]), vvec[h], u);

  float logit = 10.f * tanhf(u);
  if (maskws[b * NN + t] != 0) logit = -__builtin_inff();
  sh_logit[t] = logit;

  // deterministic counter-based Gumbel-max (stands in for jax threefry)
  const uint64_t hsh =
      splitmix64(((uint64_t)step << 40) ^ ((uint64_t)b << 16) ^
                 (uint64_t)t ^ 0x0C0FFEE123456789ULL);
  float uu = (float)((hsh >> 11) | 1ULL) * (float)(1.0 / 9007199254740992.0);
  uu = fminf(uu, 0.99999994f);
  const float g = -logf(-logf(uu));
  sh_red[t] = logit + g;
  sh_arg[t] = t;
  __syncthreads();
  for (int s = NN / 2; s > 0; s >>= 1) {
    if (t < s && sh_red[t + s] > sh_red[t]) {
      sh_red[t] = sh_red[t + s];
      sh_arg[t] = sh_arg[t + s];
    }
    __syncthreads();
  }
  if (t == 0) {
    int idx = sh_arg[0];
    if (step == 0 && idptr[0] == 0) idx = 0;   // forced first city when id==0
    s_sel = idx;
  }
  __syncthreads();
  const int idx = s_sel;

  // log-softmax at idx: max then sum-exp
  sh_red[t] = logit;
  __syncthreads();
  for (int s = NN / 2; s > 0; s >>= 1) {
    if (t < s) sh_red[t] = fmaxf(sh_red[t], sh_red[t + s]);
    __syncthreads();
  }
  if (t == 0) s_max = sh_red[0];
  __syncthreads();
  sh_red[t] = expf(logit - s_max);   // exp(-inf)=0 for masked
  __syncthreads();
  for (int s = NN / 2; s > 0; s >>= 1) {
    if (t < s) sh_red[t] += sh_red[t + s];
    __syncthreads();
  }
  if (t == 0) {
    const float logp = sh_logit[idx] - s_max - logf(sh_red[0]);
    out[OUT_LOGP + b * NN + step] = logp;
    out[OUT_IDX  + b * NN + step] = (float)idx;
    const int cprev = cur_idx[b];
    const float dx = onode[((size_t)b * NN + idx) * 2 + 0] -
                     onode[((size_t)b * NN + cprev) * 2 + 0];
    const float dy = onode[((size_t)b * NN + idx) * 2 + 1] -
                     onode[((size_t)b * NN + cprev) * 2 + 1];
    out[OUT_RS + b * NN + step] = sqrtf(dx * dx + dy * dy);
    cur_idx[b] = idx;
    maskws[b * NN + idx] = 1;
  }
  // gather h = lcv[b, idx]; latch init_h on step 0
  for (int e = t; e < EE; e += NN) {
    const float hv = lcv[((size_t)b * NN + idx) * EE + e];
    h_cur[(size_t)b * EE + e] = hv;
    if (step == 0) init_h[(size_t)b * EE + e] = hv;
  }
}

__global__ __launch_bounds__(BB)
void finalize(const float* __restrict__ onode, const int* __restrict__ cur_idx,
              float* __restrict__ out) {
  const int b = threadIdx.x;
  const int idx = cur_idx[b];
  out[OUT_LASTN + b * 2 + 0] = onode[((size_t)b * NN + idx) * 2 + 0];
  out[OUT_LASTN + b * 2 + 1] = onode[((size_t)b * NN + idx) * 2 + 1];
}

extern "C" void kernel_launch(void* const* d_in, const int* in_sizes, int n_in,
                              void* d_out, int out_size, void* d_ws, size_t ws_size,
                              hipStream_t stream) {
  const float* lcv    = (const float*)d_in[0];
  const float* onode  = (const float*)d_in[1];
  const int*   mask   = (const int*)d_in[2];
  const int*   idp    = (const int*)d_in[3];
  const float* liw    = (const float*)d_in[4];
  const float* W_hbar = (const float*)d_in[5];
  const float* b_hbar = (const float*)d_in[6];
  const float* W_rest = (const float*)d_in[7];
  const float* b_rest = (const float*)d_in[8];
  const float* W_k    = (const float*)d_in[9];
  const float* b_k    = (const float*)d_in[10];
  const float* W_q    = (const float*)d_in[11];
  const float* b_q    = (const float*)d_in[12];
  const float* vv     = (const float*)d_in[13];
  float* out = (float*)d_out;

  // workspace carve-up (~35 MB)
  float* ws = (float*)d_ws;
  size_t off = 0;
  float* keysT  = ws + off; off += (size_t)BB * NN * HH;  // keys transposed (B,H,N)
  float* mlcv   = ws + off; off += (size_t)BB * EE;
  float* h_bar  = ws + off; off += (size_t)BB * EE;
  float* rest0  = ws + off; off += (size_t)EE;
  float* query  = ws + off; off += (size_t)BB * EE;
  float* qbuf   = ws + off; off += (size_t)BB * HH;
  float* init_h = ws + off; off += (size_t)BB * EE;
  float* h_cur  = ws + off; off += (size_t)BB * EE;
  int*   maskws = (int*)(ws + off); off += (size_t)BB * NN;
  int*   curidx = (int*)(ws + off); off += (size_t)BB;
  (void)ws_size; (void)in_sizes; (void)n_in; (void)out_size;

  // phase 0: parallel precompute
  mean_lcv<<<BB * EE / 256, 256, 0, stream>>>(lcv, mlcv);
  rest0_kernel<<<1, EE, 0, stream>>>(liw, W_rest, b_rest, rest0);
  // keys^T = (lcv @ W_k + b_k) stored as (B,H,N) for coalesced per-step reads
  wmma_gemm<<<dim3(BB * NN / TM, HH / TN), 128, 0, stream>>>(
      lcv, EE, nullptr, 0, W_k, HH, b_k, nullptr, keysT, NN);
  // h_bar = mean(lcv) @ W_hbar + b_hbar
  wmma_gemm<<<dim3(BB / TM, EE / TN), 128, 0, stream>>>(
      mlcv, EE, nullptr, 0, W_hbar, EE, b_hbar, nullptr, h_bar, 0);
  init_state<<<BB * EE / 256, 256, 0, stream>>>(
      h_bar, rest0, mask, onode, query, maskws, curidx, init_h, h_cur, out);

  // phase 1: sequential decode scan (N steps, 3 kernels/step on one stream)
  for (int step = 0; step < NN; ++step) {
    // q = query @ W_q + b_q
    wmma_gemm<<<dim3(BB / TM, HH / TN), 128, 0, stream>>>(
        query, EE, nullptr, 0, W_q, HH, b_q, nullptr, qbuf, 0);
    attn_sample<<<BB, NN, 0, stream>>>(
        keysT, qbuf, vv, lcv, onode, idp, maskws, curidx, h_cur, init_h, out, step);
    // query = h_bar + concat(init_h, h_cur) @ W_rest + b_rest   (K = 512 split GEMM)
    wmma_gemm<<<dim3(BB / TM, EE / TN), 128, 0, stream>>>(
        init_h, EE, h_cur, EE, W_rest, EE, b_rest, h_bar, query, 0);
  }
  finalize<<<1, BB, 0, stream>>>(onode, curidx, out);
}
